// AgentPolicy_55903294324917
// MI455X (gfx1250) — matrix operand order repaired and checked by reading, not compiled, not run on hardware
//
#include <hip/hip_runtime.h>
#include <hip/hip_bf16.h>

// ---------------------------------------------------------------------------
// Types for CDNA5 WMMA (wave32): v_wmma_f32_16x16x32_f16
// ---------------------------------------------------------------------------
typedef __attribute__((ext_vector_type(16))) _Float16 v16h;
typedef __attribute__((ext_vector_type(8)))  _Float16 v8h;
typedef __attribute__((ext_vector_type(8)))  float    v8f;

union H8  { v8h  v; _Float16 e[8]; };
union H16 { v16h v; v8h h[2]; };

__device__ __forceinline__ v8f wmma_f16(v16h a, v16h b, v8f c) {
  // (neg_a, A, neg_b, B, c_mod, C, reuse_a, reuse_b)
  return __builtin_amdgcn_wmma_f32_16x16x32_f16(false, a, false, b, (short)0, c,
                                                false, false);
}

// A-matrix 16x32 f16 fragment from LDS (row-major, stride ld halves).
// ISA 7.12.2: lanes 0-15 -> M=lane, K groups [0..7] and [16..23];
//             lanes 16-31 -> M=lane-16, K groups [8..15] and [24..31].
__device__ __forceinline__ v16h load_a_frag_lds(const _Float16* base, int ld,
                                                int m0, int k0, int lane) {
  int m  = m0 + (lane & 15);
  int kb = (lane >> 4) << 3;
  const _Float16* p = base + m * ld + k0 + kb;
  H16 u;
  u.h[0] = *(const v8h*)(p);
  u.h[1] = *(const v8h*)(p + 16);
  return u.v;
}

// ---------------------------------------------------------------------------
// Pack f32 weight W[K,N] (row-major, in x out) into f16 B-fragments.
// Fragment (ntile,ktile) holds B = W[ktile*32 .. +31][ntile*16 .. +15].
// Lane layout: lane<16 -> col=lane, K=0..7,16..23; lane>=16 -> col=lane-16, K=8..15,24..31.
// Linear: Wp[frag*512 + lane*16 + h], frag = ntile*(K/32) + ktile.
// ---------------------------------------------------------------------------
__global__ void pack_w_kernel(const float* __restrict__ W,
                              _Float16* __restrict__ Wp, int K, int N) {
  int idx = blockIdx.x * 256 + threadIdx.x;
  if (idx >= K * N) return;
  int h    = idx & 15;
  int lane = (idx >> 4) & 31;
  int frag = idx >> 9;
  int KT   = K >> 5;
  int ntile = frag / KT;
  int ktile = frag - ntile * KT;
  int n = (ntile << 4) + (lane & 15);
  int k = (ktile << 5) + ((h >> 3) << 4) + ((lane >> 4) << 3) + (h & 7);
  Wp[idx] = (_Float16)W[k * N + n];
}

// ---------------------------------------------------------------------------
// Generic C = act(A @ W + bias). A: f32 [M,K]; W packed f16 frags; C: f32.
// Block tile 64x64, 128 threads (4 waves). Wave w owns m-tile w, 4 n-tiles,
// reusing its A fragment across all 4 WMMAs per K-step.
// M % 64 == 0, K % 32 == 0, N % 64 == 0 (true for all layers here).
// ---------------------------------------------------------------------------
template <bool RELU>
__global__ __launch_bounds__(128)
void gemm_bias_kernel(const float* __restrict__ A,
                      const _Float16* __restrict__ Wp,
                      const float* __restrict__ bias,
                      float* __restrict__ C, int M, int K, int N) {
  constexpr int LDA = 40;                 // 32 + 8 halves pad (bank spread)
  __shared__ _Float16 As[64 * LDA];
  const int tid  = threadIdx.x;
  const int lane = tid & 31;
  const int wave = tid >> 5;
  const int mbase = blockIdx.y * 64;
  const int nblk  = blockIdx.x;           // 64 columns
  const int KT = K >> 5;

  const v8f zero = {0.f, 0.f, 0.f, 0.f, 0.f, 0.f, 0.f, 0.f};
  v8f acc[4];
#pragma unroll
  for (int i = 0; i < 4; ++i) acc[i] = zero;

  const int r = tid >> 1;                 // staging row 0..63
  const int c = (tid & 1) << 4;           // staging col 0 or 16

  for (int k0 = 0; k0 < K; k0 += 32) {
    // stage A[64x32] -> LDS f16
    const float* ap = A + (size_t)(mbase + r) * K + k0 + c;
    float4 f0 = *(const float4*)(ap + 0);
    float4 f1 = *(const float4*)(ap + 4);
    float4 f2 = *(const float4*)(ap + 8);
    float4 f3 = *(const float4*)(ap + 12);
    H8 lo, hi;
    lo.e[0] = (_Float16)f0.x; lo.e[1] = (_Float16)f0.y;
    lo.e[2] = (_Float16)f0.z; lo.e[3] = (_Float16)f0.w;
    lo.e[4] = (_Float16)f1.x; lo.e[5] = (_Float16)f1.y;
    lo.e[6] = (_Float16)f1.z; lo.e[7] = (_Float16)f1.w;
    hi.e[0] = (_Float16)f2.x; hi.e[1] = (_Float16)f2.y;
    hi.e[2] = (_Float16)f2.z; hi.e[3] = (_Float16)f2.w;
    hi.e[4] = (_Float16)f3.x; hi.e[5] = (_Float16)f3.y;
    hi.e[6] = (_Float16)f3.z; hi.e[7] = (_Float16)f3.w;
    _Float16* dp = As + r * LDA + c;
    *(v8h*)(dp) = lo.v;
    *(v8h*)(dp + 8) = hi.v;
    __syncthreads();

    const v16h a = load_a_frag_lds(As, LDA, wave << 4, 0, lane);
    const int kt = k0 >> 5;
#pragma unroll
    for (int nt = 0; nt < 4; ++nt) {
      const int ntg = (nblk << 2) + nt;
      const v16h b = *(const v16h*)(Wp + ((size_t)(ntg * KT + kt) << 9) + (lane << 4));
      acc[nt] = wmma_f16(a, b, acc[nt]);
    }
    __syncthreads();
  }

  // epilogue: C/D layout -> M = r + 8*(lane>=16), N = lane&15
  const int mrow = mbase + (wave << 4) + ((lane >> 4) << 3);
#pragma unroll
  for (int nt = 0; nt < 4; ++nt) {
    const int col = (nblk << 6) + (nt << 4) + (lane & 15);
    const float bb = bias[col];
#pragma unroll
    for (int rr = 0; rr < 8; ++rr) {
      float v = acc[nt][rr] + bb;
      if (RELU) v = fmaxf(v, 0.0f);
      C[(size_t)(mrow + rr) * N + col] = v;
    }
  }
}

// ---------------------------------------------------------------------------
// Per-agent sparse attention. One workgroup (256 threads, 8 waves) per agent.
// tok = [x_row; messages] (64x256) in LDS f16.
// k = tok@Wk+bk (WMMA, 64 tiles), q rows 0..15 = tok@Wq+bq (WMMA, 16 tiles).
// Block scores (query block 0), top-2-of-4 mask, softmax row 0,
// z = attn_row @ tok, out = z@Wv+bv, comm = out@Wo+bo, xr = x + comm.
// ---------------------------------------------------------------------------
__global__ __launch_bounds__(256)
void attn_kernel(const float* __restrict__ xin,        // [B,256]
                 const float* __restrict__ messages,   // [B,63,256]
                 const _Float16* __restrict__ Wqp, const float* __restrict__ bq,
                 const _Float16* __restrict__ Wkp, const float* __restrict__ bk,
                 const float* __restrict__ Wv, const float* __restrict__ bv,
                 const float* __restrict__ Wo, const float* __restrict__ bo,
                 float* __restrict__ xr) {
  constexpr int LDT = 264;                // 256 + 8 halves pad
  extern __shared__ char smemraw[];
  _Float16* tok = (_Float16*)smemraw;      // 64*LDT
  _Float16* ksm = tok + 64 * LDT;          // 64*LDT
  _Float16* qsm = ksm + 64 * LDT;          // 16*LDT
  float* fbuf  = (float*)(qsm + 16 * LDT);
  float* srow  = fbuf;                     // 64
  float* attnw = fbuf + 64;                // 64
  float* red   = fbuf + 128;               // 32
  float* sblkS = fbuf + 160;               // 4
  float* zs    = fbuf + 164;               // 256
  float* os    = fbuf + 420;               // 256

  const int b    = blockIdx.x;
  const int tid  = threadIdx.x;
  const int lane = tid & 31;
  const int wave = tid >> 5;

  // ---- 1) stage tokens (f32 -> f16) ----
  {
    const float* mrow = messages + (size_t)b * 63 * 256;
    const float* xrow = xin + (size_t)b * 256;
    tok[tid] = (_Float16)xrow[tid];
    for (int rr = 1; rr < 64; ++rr)
      tok[rr * LDT + tid] = (_Float16)mrow[(rr - 1) * 256 + tid];
  }
  __syncthreads();

  const v8f zero = {0.f, 0.f, 0.f, 0.f, 0.f, 0.f, 0.f, 0.f};

  // ---- 2) K projection: 4 m-tiles x 16 n-tiles over 8 waves ----
  {
    const int mt  = wave & 3;
    const int nb0 = (wave >> 2) << 3;      // 0 or 8
    v8f acc[8];
#pragma unroll
    for (int i = 0; i < 8; ++i) acc[i] = zero;
#pragma unroll
    for (int kk = 0; kk < 8; ++kk) {
      const v16h a = load_a_frag_lds(tok, LDT, mt << 4, kk << 5, lane);
#pragma unroll
      for (int nt = 0; nt < 8; ++nt) {
        const v16h bm = *(const v16h*)(Wkp + ((size_t)(((nb0 + nt) << 3) + kk) << 9) + (lane << 4));
        acc[nt] = wmma_f16(a, bm, acc[nt]);
      }
    }
    const int rbase = (mt << 4) + ((lane >> 4) << 3);
#pragma unroll
    for (int nt = 0; nt < 8; ++nt) {
      const int col = ((nb0 + nt) << 4) + (lane & 15);
      const float bb = bk[col];
#pragma unroll
      for (int rr = 0; rr < 8; ++rr)
        ksm[(rbase + rr) * LDT + col] = (_Float16)(acc[nt][rr] + bb);
    }
  }

  // ---- 3) Q projection rows 0..15: 16 n-tiles over 8 waves (2 each) ----
  {
    v8f acc[2];
    acc[0] = zero; acc[1] = zero;
#pragma unroll
    for (int kk = 0; kk < 8; ++kk) {
      const v16h a = load_a_frag_lds(tok, LDT, 0, kk << 5, lane);
#pragma unroll
      for (int j = 0; j < 2; ++j) {
        const int ntg = (wave << 1) + j;
        const v16h bm = *(const v16h*)(Wqp + ((size_t)((ntg << 3) + kk) << 9) + (lane << 4));
        acc[j] = wmma_f16(a, bm, acc[j]);
      }
    }
    const int rbase = (lane >> 4) << 3;
#pragma unroll
    for (int j = 0; j < 2; ++j) {
      const int col = (((wave << 1) + j) << 4) + (lane & 15);
      const float bb = bq[col];
#pragma unroll
      for (int rr = 0; rr < 8; ++rr)
        qsm[(rbase + rr) * LDT + col] = (_Float16)(acc[j][rr] + bb);
    }
  }
  __syncthreads();

  // ---- 4) block scores for query block 0: sblk[m] = mean(q[0:16]) . mean(k[16m:16m+16])
  {
    float qb = 0.f;
#pragma unroll
    for (int rr = 0; rr < 16; ++rr) qb += (float)qsm[rr * LDT + tid];
    qb *= (1.0f / 16.0f);
#pragma unroll
    for (int m = 0; m < 4; ++m) {
      float kbv = 0.f;
#pragma unroll
      for (int rr = 0; rr < 16; ++rr) kbv += (float)ksm[(m * 16 + rr) * LDT + tid];
      kbv *= (1.0f / 16.0f);
      float p = qb * kbv;
      p += __shfl_xor(p, 16, 32);
      p += __shfl_xor(p, 8, 32);
      p += __shfl_xor(p, 4, 32);
      p += __shfl_xor(p, 2, 32);
      p += __shfl_xor(p, 1, 32);
      if (lane == 0) red[wave * 4 + m] = p;
    }
  }
  __syncthreads();
  if (tid < 4) {
    float s = 0.f;
    for (int w = 0; w < 8; ++w) s += red[w * 4 + tid];
    sblkS[tid] = s;
  }
  __syncthreads();

  // ---- 5) masked scores, row 0 ----
  if (tid < 64) {
    float s = 0.f;
    for (int d = 0; d < 256; ++d)
      s += (float)qsm[d] * (float)ksm[tid * LDT + d];
    s *= (1.0f / 16.0f);                      // 1/sqrt(256)
    // top-2-of-4 threshold (keep = round(0.5*4) = 2), mask = sblk >= thr
    float mx = -1e30f, second = -1e30f;
#pragma unroll
    for (int m = 0; m < 4; ++m) {
      float v = sblkS[m];
      if (v > mx) { second = mx; mx = v; }
      else if (v > second) second = v;
    }
    if (!(sblkS[tid >> 4] >= second)) s = -1e9f;
    srow[tid] = s;
  }
  __syncthreads();
  if (tid < 64) {
    float mx = -1e30f;
    for (int t = 0; t < 64; ++t) mx = fmaxf(mx, srow[t]);
    attnw[tid] = expf(srow[tid] - mx);
  }
  __syncthreads();

  // ---- 6) z = attn_row @ tok (normalized) ----
  {
    float ssum = 0.f;
    for (int t = 0; t < 64; ++t) ssum += attnw[t];
    const float inv = 1.0f / ssum;
    float z = 0.f;
    for (int t = 0; t < 64; ++t) z += attnw[t] * (float)tok[t * LDT + tid];
    zs[tid] = z * inv;
  }
  __syncthreads();

  // ---- 7) out = z @ Wv + bv (since sum(attn)=1 the bias passes through) ----
  {
    float o = bv[tid];
    for (int k = 0; k < 256; ++k) o += zs[k] * Wv[k * 256 + tid];
    os[tid] = o;
  }
  __syncthreads();

  // ---- 8) comm = out @ Wo + bo ; residual ----
  {
    float cacc = bo[tid];
    for (int k = 0; k < 256; ++k) cacc += os[k] * Wo[k * 256 + tid];
    xr[(size_t)b * 256 + tid] = xin[(size_t)b * 256 + tid] + cacc;
  }
}

// ---------------------------------------------------------------------------
// Host launch
// ---------------------------------------------------------------------------
extern "C" void kernel_launch(void* const* d_in, const int* in_sizes, int n_in,
                              void* d_out, int out_size, void* d_ws, size_t ws_size,
                              hipStream_t stream) {
  const float* local_obs = (const float*)d_in[0];   // [B,512]
  const float* messages  = (const float*)d_in[1];   // [B,63,256]
  const float* W1 = (const float*)d_in[2];  const float* b1 = (const float*)d_in[3];
  const float* W2 = (const float*)d_in[4];  const float* b2 = (const float*)d_in[5];
  const float* W3 = (const float*)d_in[6];  const float* b3 = (const float*)d_in[7];
  const float* W4 = (const float*)d_in[8];  const float* b4 = (const float*)d_in[9];
  const float* Wq = (const float*)d_in[10]; const float* bq = (const float*)d_in[11];
  const float* Wk = (const float*)d_in[12]; const float* bk = (const float*)d_in[13];
  const float* Wv = (const float*)d_in[14]; const float* bv = (const float*)d_in[15];
  const float* Wo = (const float*)d_in[16]; const float* bo = (const float*)d_in[17];

  const int B = in_sizes[0] / 512;                  // 4096

  char* ws = (char*)d_ws;
  size_t off = 0;
  auto alloc = [&](size_t bytes) -> char* {
    off = (off + 255) & ~(size_t)255;
    char* p = ws + off;
    off += bytes;
    return p;
  };

  _Float16* W1p = (_Float16*)alloc((size_t)512 * 128 * 2);
  _Float16* W2p = (_Float16*)alloc((size_t)128 * 256 * 2);
  _Float16* Wqp = (_Float16*)alloc((size_t)256 * 256 * 2);
  _Float16* Wkp = (_Float16*)alloc((size_t)256 * 256 * 2);
  _Float16* W3p = (_Float16*)alloc((size_t)256 * 128 * 2);
  _Float16* W4p = (_Float16*)alloc((size_t)128 * 256 * 2);
  float* h1   = (float*)alloc((size_t)B * 128 * 4);
  float* xbuf = (float*)alloc((size_t)B * 256 * 4);
  float* xres = (float*)alloc((size_t)B * 256 * 4);
  float* h3   = (float*)alloc((size_t)B * 128 * 4);

  pack_w_kernel<<<(512 * 128) / 256, 256, 0, stream>>>(W1, W1p, 512, 128);
  pack_w_kernel<<<(128 * 256) / 256, 256, 0, stream>>>(W2, W2p, 128, 256);
  pack_w_kernel<<<(256 * 256) / 256, 256, 0, stream>>>(Wq, Wqp, 256, 256);
  pack_w_kernel<<<(256 * 256) / 256, 256, 0, stream>>>(Wk, Wkp, 256, 256);
  pack_w_kernel<<<(256 * 128) / 256, 256, 0, stream>>>(W3, W3p, 256, 128);
  pack_w_kernel<<<(128 * 256) / 256, 256, 0, stream>>>(W4, W4p, 128, 256);

  // encoder
  gemm_bias_kernel<true><<<dim3(128 / 64, B / 64), 128, 0, stream>>>(
      local_obs, W1p, b1, h1, B, 512, 128);
  gemm_bias_kernel<true><<<dim3(256 / 64, B / 64), 128, 0, stream>>>(
      h1, W2p, b2, xbuf, B, 128, 256);

  // sparse attention + residual
  constexpr int LDT = 264;
  const size_t smem = (size_t)(64 + 64 + 16) * LDT * sizeof(_Float16) +
                      (size_t)(64 + 64 + 32 + 4 + 256 + 256) * sizeof(float);
  attn_kernel<<<B, 256, smem, stream>>>(xbuf, messages, Wqp, bq, Wkp, bk,
                                        Wv, bv, Wo, bo, xres);

  // decoder
  gemm_bias_kernel<true><<<dim3(128 / 64, B / 64), 128, 0, stream>>>(
      xres, W3p, b3, h3, B, 256, 128);
  gemm_bias_kernel<false><<<dim3(256 / 64, B / 64), 128, 0, stream>>>(
      h3, W4p, b4, (float*)d_out, B, 128, 256);
}
